// ST_ATT_Layer_34703335751714
// MI455X (gfx1250) — matrix operand order repaired and checked by reading, not compiled, not run on hardware
//
#include <hip/hip_runtime.h>

typedef _Float16 h16;
typedef __attribute__((ext_vector_type(16))) _Float16 v16h;
typedef __attribute__((ext_vector_type(8)))  _Float16 v8h;
typedef __attribute__((ext_vector_type(4)))  _Float16 v4h;
typedef __attribute__((ext_vector_type(8)))  float    v8f;
typedef __attribute__((ext_vector_type(4)))  float    v4f;

#define BATCH  512
#define SEQ    176
#define DM     256
#define HEADS  8
#define HDIM   32
#define NJ     22      // joints per frame
#define FPB    4       // frames per block
#define ROWS   (FPB*NJ)   // 88 valid rows per block
#define MTILES 6          // 96 padded rows
#define XS     264        // X LDS row stride (halves), 528B = 33*16
#define QKS    264        // Q/K LDS row stride (halves)
#define QKROWS 136        // 128 + 8 spill rows so stores need no bounds check
#define VTS    136        // Vt row stride (halves), 272B = 17*16

// halves offsets into dynamic LDS (all 16B aligned)
#define OFF_X  0
#define OFF_Q  (96*XS)                  // 25344
#define OFF_K  (OFF_Q + QKROWS*QKS)     // 61248
#define OFF_VT (OFF_K + QKROWS*QKS)     // 97152
#define OFF_P  (OFF_VT + 256*VTS)       // 131968
#define SMEM_H (OFF_P + 8*32*32)        // 140160 halves = 280320 B

static __device__ inline v8f wmma16(v16h a, v16h b, v8f c) {
  return __builtin_amdgcn_wmma_f32_16x16x32_f16(false, a, false, b, (short)0, c,
                                                false, false);
}

// A fragment (16x32 f16, M rows x K cols) from row-major src.
static __device__ inline v16h load_frag_a(const h16* src, int stride, int row0,
                                          int col0, int lane) {
  int li = lane & 15, g = lane >> 4;
  const h16* p = src + (size_t)(row0 + li) * stride + col0 + g * 8;
  union { v16h v; v8h h[2]; } u;
  u.h[0] = *(const v8h*)p;
  u.h[1] = *(const v8h*)(p + 16);
  return u.v;
}

// B fragment (32x16 f16): B[k,n] = src[row0+n][col0+k], src row-major along k.
static __device__ inline v16h load_frag_b(const h16* src, int stride, int row0,
                                          int col0, int lane) {
  int nn = lane & 15, g = lane >> 4;
  const h16* p = src + (size_t)(row0 + nn) * stride + col0 + g * 8;
  union { v16h v; v8h h[2]; } u;
  u.h[0] = *(const v8h*)p;
  u.h[1] = *(const v8h*)(p + 16);
  return u.v;
}

__global__ void convert_weights(const float* __restrict__ wq,
                                const float* __restrict__ wk,
                                const float* __restrict__ wv,
                                h16* __restrict__ wh) {
  int i = blockIdx.x * 256 + threadIdx.x;   // grid 256 -> 65536 exactly
  wh[i]          = (h16)wq[i];
  wh[i + 65536]  = (h16)wk[i];
  wh[i + 131072] = (h16)wv[i];
}

__global__ __launch_bounds__(256, 1)
void st_att_fused(const float* __restrict__ x,
                  const float* __restrict__ bq,
                  const float* __restrict__ bk,
                  const float* __restrict__ bv,
                  const h16*  __restrict__ wh,
                  float* __restrict__ out) {
  extern __shared__ h16 sm[];
  h16* Xs = sm + OFF_X;    // [96][XS]  compact rows
  h16* Qs = sm + OFF_Q;    // [136][QKS] frame-strided (32 rows/frame) + spill
  h16* Ks = sm + OFF_K;    // [136][QKS]
  h16* Vt = sm + OFF_VT;   // [256][VTS] transposed, frame-strided rows
  h16* Ps = sm + OFF_P;    // [8 waves][32][32]

  const int tid  = threadIdx.x;
  const int lane = tid & 31;
  const int wave = __builtin_amdgcn_readfirstlane(tid >> 5);  // scalar!
  const int b     = blockIdx.x >> 1;
  const int fbase = (blockIdx.x & 1) * FPB;

  // ---- zero Vt (so K-dim padding contributes exactly 0 in P@V) ----
  {
    v8h z = {(h16)0,(h16)0,(h16)0,(h16)0,(h16)0,(h16)0,(h16)0,(h16)0};
    v8h* vt8 = (v8h*)Vt;
    for (int i = tid; i < 256 * VTS / 8; i += 256) vt8[i] = z;
  }

  // ---- stage x + positional embedding -> f16 LDS ----
  for (int it = tid; it < ROWS * 64; it += 256) {
    int r   = it >> 6;
    int c4  = (it & 63) << 2;
    int srow = fbase * NJ + r;
    float pos = (float)(r % NJ);
    v4f xv = *(const v4f*)(x + ((size_t)b * SEQ + srow) * DM + c4);
    v4h hv;
#pragma unroll
    for (int u = 0; u < 4; ++u) {
      int col = c4 + u;
      // div = exp(-(col&~1) * ln(10000)/256)
      float arg = pos * __expf((float)(col & ~1) * -0.0359778921f);
      float pe  = (col & 1) ? __cosf(arg) : __sinf(arg);
      hv[u] = (h16)(xv[u] + pe);
    }
    *(v4h*)(Xs + (size_t)r * XS + c4) = hv;
  }
  __syncthreads();

  const int li = lane & 15, g = lane >> 4;

  // ---- fused QKV projection: Y[96,256] = X[96,256] @ W^T, per wave 6 N-tiles ----
  for (int nt = wave; nt < 48; nt += 8) {            // nt is scalar
    const int which = nt >> 4;                       // 0=Q 1=K 2=V (scalar)
    const int ncol  = (nt & 15) << 4;                // scalar
    const h16* W = wh + which * 65536;
    v16h bf[8];
#pragma unroll
    for (int ks = 0; ks < 8; ++ks)
      bf[ks] = load_frag_b(W, DM, ncol, ks * 32, lane);
    const float* bias = (which == 0) ? bq : ((which == 1) ? bk : bv);
    float bval = bias[ncol + li];
#pragma unroll
    for (int mt = 0; mt < MTILES; ++mt) {
      v8f acc;
#pragma unroll
      for (int u = 0; u < 8; ++u) acc[u] = bval;
#pragma unroll
      for (int ks = 0; ks < 8; ++ks) {
        v16h a = load_frag_a(Xs, XS, mt * 16, ks * 32, lane);
        acc = wmma16(a, bf[ks], acc);
      }
      // unconditional stores: compact row r -> frame-strided row rr;
      // spill rows (r>=88 -> rr in [128,136)) land in-bounds and are never read
      if (which == 2) {
        h16* dv = Vt + (size_t)(ncol + li) * VTS;
#pragma unroll
        for (int u = 0; u < 8; ++u) {
          int r  = mt * 16 + u + g * 8;              // two values per u (g sel)
          int rr = (r / NJ) * 32 + (r % NJ);
          dv[rr] = (h16)acc[u];
        }
      } else {
        h16* d = ((which == 0) ? Qs : Ks) + ncol + li;   // scalar select
#pragma unroll
        for (int u = 0; u < 8; ++u) {
          int r  = mt * 16 + u + g * 8;
          int rr = (r / NJ) * 32 + (r % NJ);
          d[(size_t)rr * QKS] = (h16)acc[u];
        }
      }
    }
  }
  __syncthreads();

  // ---- attention: each wave owns 4 (head, frame) pairs ----
  const float scale = 0.17677669529663687f;   // 1/sqrt(32)
  h16* Pw = Ps + wave * 32 * 32;
  const bool v1 = li < (NJ - 16);             // valid key in tile-1

  for (int pi = 0; pi < 4; ++pi) {
    const int pair = wave * 4 + pi;           // scalar
    const int h  = pair >> 2;
    const int fl = pair & 3;
    const int rq = fl * 32;                   // frame-strided base row
    const int ch = h * HDIM;

    // scores S = Q Kt / sqrt(d)
    v16h kb0 = load_frag_b(Ks, QKS, rq,      ch, lane);
    v16h kb1 = load_frag_b(Ks, QKS, rq + 16, ch, lane);
    v8f s[2][2];
#pragma unroll
    for (int mt = 0; mt < 2; ++mt) {
      v16h aq = load_frag_a(Qs, QKS, rq + mt * 16, ch, lane);
      v8f z = {0.f,0.f,0.f,0.f,0.f,0.f,0.f,0.f};
      s[mt][0] = wmma16(aq, kb0, z);
      s[mt][1] = wmma16(aq, kb1, z);
    }

    // masked softmax over 22 keys; a row lives in one 16-lane half
#pragma unroll
    for (int mt = 0; mt < 2; ++mt) {
#pragma unroll
      for (int u = 0; u < 8; ++u) {
        float s0 = s[mt][0][u] * scale;
        float s1 = v1 ? s[mt][1][u] * scale : -1e30f;
        float m = fmaxf(s0, s1);
#pragma unroll
        for (int msk = 1; msk <= 8; msk <<= 1)
          m = fmaxf(m, __shfl_xor(m, msk, 32));
        float e0 = __expf(s0 - m);
        float e1 = __expf(s1 - m);
        float t = e0 + e1;
#pragma unroll
        for (int msk = 1; msk <= 8; msk <<= 1)
          t += __shfl_xor(t, msk, 32);
        float rd = 1.0f / t;
        int row = mt * 16 + u + g * 8;
        Pw[row * 32 + li]      = (h16)(e0 * rd);
        Pw[row * 32 + 16 + li] = (h16)(e1 * rd);
      }
    }

    // O = P @ V  (K padded to 32; P padding is exact zeros, Vt zero-filled)
    v16h vb0 = load_frag_b(Vt, VTS, ch,      rq, lane);
    v16h vb1 = load_frag_b(Vt, VTS, ch + 16, rq, lane);
    const size_t obase0 = ((size_t)b * SEQ + (size_t)(fbase + fl) * NJ) * DM + ch + li;

    {   // mt = 0: rows 0..15 all valid -> unconditional stores
      v16h ap = load_frag_a(Pw, 32, 0, 0, lane);
      v8f z = {0.f,0.f,0.f,0.f,0.f,0.f,0.f,0.f};
      v8f o0 = wmma16(ap, vb0, z);
      v8f o1 = wmma16(ap, vb1, z);
      size_t base = obase0 + (size_t)(g * 8) * DM;
#pragma unroll
      for (int u = 0; u < 8; ++u) {
        out[base + (size_t)u * DM]      = o0[u];
        out[base + (size_t)u * DM + 16] = o1[u];
      }
    }
    {   // mt = 1: only rows 16..21 valid -> g==0, u<6
      v16h ap = load_frag_a(Pw, 32, 16, 0, lane);
      v8f z = {0.f,0.f,0.f,0.f,0.f,0.f,0.f,0.f};
      v8f o0 = wmma16(ap, vb0, z);
      v8f o1 = wmma16(ap, vb1, z);
      if (g == 0) {
        size_t base = obase0 + (size_t)16 * DM;
#pragma unroll
        for (int u = 0; u < 6; ++u) {
          out[base + (size_t)u * DM]      = o0[u];
          out[base + (size_t)u * DM + 16] = o1[u];
        }
      }
    }
  }
}

extern "C" void kernel_launch(void* const* d_in, const int* in_sizes, int n_in,
                              void* d_out, int out_size, void* d_ws, size_t ws_size,
                              hipStream_t stream) {
  const float* x  = (const float*)d_in[0];
  const float* wq = (const float*)d_in[1];
  const float* bq = (const float*)d_in[2];
  const float* wk = (const float*)d_in[3];
  const float* bk = (const float*)d_in[4];
  const float* wv = (const float*)d_in[5];
  const float* bv = (const float*)d_in[6];
  float* out = (float*)d_out;
  h16* wh = (h16*)d_ws;   // 3 * 256*256 f16 = 384 KB

  convert_weights<<<256, 256, 0, stream>>>(wq, wk, wv, wh);

  size_t smem = (size_t)SMEM_H * sizeof(h16);   // ~274 KB
  hipFuncSetAttribute((const void*)st_att_fused,
                      hipFuncAttributeMaxDynamicSharedMemorySize, (int)smem);
  st_att_fused<<<BATCH * 2, 256, smem, stream>>>(x, bq, bk, bv, wh, out);
}
